// LSTM_67611375174246
// MI455X (gfx1250) — matrix operand order repaired and checked by reading, not compiled, not run on hardware
//
#include <hip/hip_runtime.h>
#include <hip/hip_bf16.h>

typedef __attribute__((ext_vector_type(16))) __bf16        v16bf;
typedef __attribute__((ext_vector_type(8)))  float         v8f;
typedef __attribute__((ext_vector_type(4)))  unsigned int  v4u;

#define B_     128
#define T_     512
#define DIN    256
#define DH     512
#define DOUT   256
#define KTOT   768              // DIN + DH
#define NWG    32               // 32 workgroups, each owns 16 h-columns
#define NBLK_THREADS 256        // 8 wave32 per WG

union FragU { v16bf v; v4u q[2]; unsigned short us[16]; };

__device__ inline unsigned short f2bf(float f) {
    unsigned int u = __builtin_bit_cast(unsigned int, f);
    unsigned int r = u + 0x7FFFu + ((u >> 16) & 1u);   // round-to-nearest-even
    return (unsigned short)(r >> 16);
}

__device__ inline float sigmoid_f(float x) { return 1.0f / (1.0f + __expf(-x)); }
__device__ inline float tanh_f(float x)    { float e = __expf(-2.0f * x); return (1.0f - e) / (1.0f + e); }

// A-fragment (16x32 bf16, M x K), per CDNA5 ISA layout:
// lane l: row m = l%16; halves: lanes<16 hold K = kb+[0..7] & kb+16+[0..7],
//                       lanes>=16 hold K = kb+8+[0..7] & kb+24+[0..7].
__device__ inline v16bf load_a_frag(const unsigned short* xrow, const unsigned short* hrow,
                                    int kb, int lane) {
    const int off = (lane < 16) ? 0 : 8;
    const int k0 = kb + off;
    const int k1 = kb + 16 + off;
    FragU u;
    const unsigned short* s0 = (k0 < DIN) ? (xrow + k0) : (hrow + (k0 - DIN));
    const unsigned short* s1 = (k1 < DIN) ? (xrow + k1) : (hrow + (k1 - DIN));
    u.q[0] = *(const v4u*)s0;
    u.q[1] = *(const v4u*)s1;
    return u.v;
}

// B-fragment (32x16 bf16, K x N) from LDS stored col-major: wl[col*KTOT + k].
// lane l: col n = l%16; lanes<16 hold K = kb+[0..7] & kb+16+[0..7], lanes>=16 hold K = kb+8+[0..7] & kb+24+[0..7].
__device__ inline v16bf load_b_frag(const unsigned short* wl, int col_local, int kb, int lane) {
    const int ks = kb + ((lane < 16) ? 0 : 8);
    const unsigned short* p = wl + (size_t)col_local * KTOT + ks;
    FragU u;
    u.q[0] = *(const v4u*)p;
    u.q[1] = *(const v4u*)(p + 16);
    return u.v;
}

__device__ inline void grid_barrier(int* cnt, int target) {
    __syncthreads();
    if (threadIdx.x == 0) {
        __threadfence();
        __hip_atomic_fetch_add(cnt, 1, __ATOMIC_ACQ_REL, __HIP_MEMORY_SCOPE_AGENT);
        while (__hip_atomic_load(cnt, __ATOMIC_ACQUIRE, __HIP_MEMORY_SCOPE_AGENT) < target) {
            __builtin_amdgcn_s_sleep(8);
        }
    }
    __syncthreads();
}

// ---------------------------------------------------------------------------
// Kernel 1: transpose + convert x [B,T,DIN] f32 -> xT [T,B,DIN] bf16
// ---------------------------------------------------------------------------
__global__ void k_xpose(const float* __restrict__ x, unsigned short* __restrict__ xT) {
    size_t i = (size_t)blockIdx.x * NBLK_THREADS + threadIdx.x;
    const size_t total = (size_t)B_ * T_ * DIN;
    if (i >= total) return;
    int d = (int)(i & (DIN - 1));
    size_t bt = i >> 8;                 // DIN == 256
    int t = (int)(bt & (T_ - 1));
    size_t b = bt >> 9;                 // T_ == 512
    xT[((size_t)t * B_ + b) * DIN + d] = f2bf(x[i]);
}

// ---------------------------------------------------------------------------
// Kernel 2: persistent LSTM recurrence. 32 WGs x 8 waves.
//   WG wg owns h-columns [16*wg, 16*wg+16); its 4 gate column-tiles live at
//   columns g*512 + 16*wg. Weight slice (768 x 64 bf16, col-major) in LDS.
//   Wave wv owns rows [16*wv, 16*wv+16); c-tile is register resident (v8f).
// ---------------------------------------------------------------------------
__global__ void k_lstm(const unsigned short* __restrict__ xT,
                       const float* __restrict__ w_x2h,   // [DIN, 4*DH]
                       const float* __restrict__ w_h2h,   // [DH, 4*DH]
                       const float* __restrict__ b_x2h,
                       const float* __restrict__ b_h2h,
                       unsigned short* __restrict__ hbuf0,  // [B_, DH] bf16
                       unsigned short* __restrict__ hbuf1,
                       int* __restrict__ cnt) {
    extern __shared__ unsigned short w_lds[];   // 64 cols * 768 K * 2B = 96 KB
    const int wg   = blockIdx.x;                // 0..31
    const int tid  = threadIdx.x;               // 0..255
    const int lane = tid & 31;
    const int wv   = tid >> 5;                  // 0..7

    // --- stage weight slice into LDS, col-major (converted f32 -> bf16) ---
    for (int idx = tid; idx < 64 * KTOT; idx += NBLK_THREADS) {
        const int c = idx / KTOT;               // local col 0..63  (= g*16 + cc)
        const int k = idx - c * KTOT;           // 0..767
        const int g = c >> 4;
        const int cc = c & 15;
        const int gcol = g * DH + wg * 16 + cc; // column in [0, 2048)
        const float f = (k < DIN) ? w_x2h[(size_t)k * (4 * DH) + gcol]
                                  : w_h2h[(size_t)(k - DIN) * (4 * DH) + gcol];
        w_lds[idx] = f2bf(f);
    }
    __syncthreads();

    // per-lane bias for each gate (column = lane%16 within this WG's slice)
    float bias[4];
#pragma unroll
    for (int g = 0; g < 4; ++g) {
        const int gcol = g * DH + wg * 16 + (lane & 15);
        bias[g] = b_x2h[gcol] + b_h2h[gcol];
    }

    const int rowA = wv * 16 + (lane & 15);       // A-load row for this lane
    const int colH = wg * 16 + (lane & 15);       // output h/c column
    v8f c_acc = {0.f, 0.f, 0.f, 0.f, 0.f, 0.f, 0.f, 0.f};

    int cur = 0;
    int target = NWG;
    for (int t = 0; t < T_; ++t) {
        const unsigned short* xrow = xT + ((size_t)t * B_ + rowA) * DIN;
        const unsigned short* hrow = (cur ? hbuf1 : hbuf0) + (size_t)rowA * DH;
        unsigned short* hnext = (cur ? hbuf0 : hbuf1);

        v8f acc[4];
#pragma unroll
        for (int g = 0; g < 4; ++g) acc[g] = (v8f){0.f, 0.f, 0.f, 0.f, 0.f, 0.f, 0.f, 0.f};

        for (int kb = 0; kb < KTOT; kb += 32) {
            const v16bf a = load_a_frag(xrow, hrow, kb, lane);
#pragma unroll
            for (int g = 0; g < 4; ++g) {
                const v16bf b = load_b_frag(w_lds, g * 16 + (lane & 15), kb, lane);
                acc[g] = __builtin_amdgcn_wmma_f32_16x16x32_bf16(
                    false, a, false, b, (short)0, acc[g], false, false);
            }
        }

        // elementwise gates: C/D layout -> VGPR e, lane l: row = 16*wv + e + (l<16?0:8)
#pragma unroll
        for (int e = 0; e < 8; ++e) {
            const float gi = sigmoid_f(acc[0][e] + bias[0]);
            const float gf = sigmoid_f(acc[1][e] + bias[1]);
            const float gg = tanh_f   (acc[2][e] + bias[2]);
            const float go = sigmoid_f(acc[3][e] + bias[3]);
            const float cn = c_acc[e] * gf + gi * gg;
            c_acc[e] = cn;
            const float hn = go * tanh_f(cn);
            const int row = wv * 16 + e + ((lane < 16) ? 0 : 8);
            hnext[(size_t)row * DH + colH] = f2bf(hn);
        }

        grid_barrier(cnt, target);
        target += NWG;
        cur ^= 1;
    }
}

// ---------------------------------------------------------------------------
// Kernel 3: out = h_last @ w_fc + b_fc  via WMMA. 16 WGs x 8 waves = 128 waves,
// one 16x16 tile each (8 M-tiles x 16 N-tiles).
// ---------------------------------------------------------------------------
__global__ void k_fc(const unsigned short* __restrict__ hfin,  // [B_, DH] bf16
                     const float* __restrict__ w_fc,           // [DH, DOUT]
                     const float* __restrict__ b_fc,
                     float* __restrict__ out) {                // [B_, DOUT]
    const int tid  = threadIdx.x;
    const int lane = tid & 31;
    const int wv   = tid >> 5;
    const int wglob = blockIdx.x * 8 + wv;      // 0..127
    const int mt = wglob >> 4;                  // 0..7
    const int nt = wglob & 15;                  // 0..15

    v8f acc = {0.f, 0.f, 0.f, 0.f, 0.f, 0.f, 0.f, 0.f};
    const int rowA = mt * 16 + (lane & 15);
    const int ncol = nt * 16 + (lane & 15);

    for (int kb = 0; kb < DH; kb += 32) {
        const int off = (lane < 16) ? 0 : 8;
        FragU ua;
        ua.q[0] = *(const v4u*)(hfin + (size_t)rowA * DH + kb + off);
        ua.q[1] = *(const v4u*)(hfin + (size_t)rowA * DH + kb + 16 + off);

        FragU ub;
        const int ks = kb + ((lane < 16) ? 0 : 8);
#pragma unroll
        for (int j = 0; j < 16; ++j)
            ub.us[j] = f2bf(w_fc[(size_t)(ks + (j < 8 ? j : j + 8)) * DOUT + ncol]);

        acc = __builtin_amdgcn_wmma_f32_16x16x32_bf16(
            false, ua.v, false, ub.v, (short)0, acc, false, false);
    }

    const float bias = b_fc[ncol];
#pragma unroll
    for (int e = 0; e < 8; ++e) {
        const int row = mt * 16 + e + ((lane < 16) ? 0 : 8);
        out[(size_t)row * DOUT + ncol] = acc[e] + bias;
    }
}

// ---------------------------------------------------------------------------
extern "C" void kernel_launch(void* const* d_in, const int* in_sizes, int n_in,
                              void* d_out, int out_size, void* d_ws, size_t ws_size,
                              hipStream_t stream) {
    const float* x     = (const float*)d_in[0];   // [128,512,256]
    const float* w_x2h = (const float*)d_in[1];   // [256,2048]
    const float* b_x2h = (const float*)d_in[2];   // [2048]
    const float* w_h2h = (const float*)d_in[3];   // [512,2048]
    const float* b_h2h = (const float*)d_in[4];   // [2048]
    const float* w_fc  = (const float*)d_in[5];   // [512,256]
    const float* b_fc  = (const float*)d_in[6];   // [256]
    float* out = (float*)d_out;                   // [128,256]

    const size_t XT_BYTES = (size_t)T_ * B_ * DIN * sizeof(unsigned short); // 32 MB
    const size_t H_BYTES  = (size_t)B_ * DH * sizeof(unsigned short);       // 128 KB

    unsigned short* xT = (unsigned short*)d_ws;
    unsigned short* h0 = (unsigned short*)((char*)d_ws + XT_BYTES);
    unsigned short* h1 = (unsigned short*)((char*)d_ws + XT_BYTES + H_BYTES);
    int* cnt = (int*)((char*)d_ws + XT_BYTES + 2 * H_BYTES);

    // zero h0 (h_0 = 0), h1 and the barrier counter (deterministic per launch)
    hipMemsetAsync((char*)d_ws + XT_BYTES, 0, 2 * H_BYTES + 64, stream);

    const size_t nel = (size_t)B_ * T_ * DIN;
    k_xpose<<<(unsigned)((nel + NBLK_THREADS - 1) / NBLK_THREADS), NBLK_THREADS, 0, stream>>>(x, xT);

    k_lstm<<<NWG, NBLK_THREADS, 64 * KTOT * sizeof(unsigned short), stream>>>(
        xT, w_x2h, w_h2h, b_x2h, b_h2h, h0, h1, cnt);

    // T_=512 steps: step t writes buffer (t+1)%2, so final h lives in h0.
    k_fc<<<16, NBLK_THREADS, 0, stream>>>(h0, w_fc, b_fc, out);
}
